// SimpleCNN_60335700574585
// MI455X (gfx1250) — matrix operand order repaired and checked by reading, not compiled, not run on hardware
//
#include <hip/hip_runtime.h>
#include <hip/hip_bf16.h>
#include <math.h>

// ---------------- CDNA5 WMMA types ----------------
typedef __attribute__((ext_vector_type(16))) _Float16 v16h;
typedef __attribute__((ext_vector_type(8)))  _Float16 v8h;
typedef __attribute__((ext_vector_type(8)))  float    v8f;

// A-fragment (16x32 f16, M x K): lanes 0-15 hold row M=lane, halves {K0..7, K16..23};
// lanes 16-31 hold row M=lane-16, halves {K8..15, K24..31}.  (ISA 7.12.2)
__device__ __forceinline__ v16h load_a_frag(const _Float16* rowbase, int ldh, int kb, int lane) {
    int r = lane & 15;
    const _Float16* p = rowbase + (size_t)r * ldh + kb + ((lane & 16) ? 8 : 0);
    v8h lo = *(const v8h*)(p);
    v8h hi = *(const v8h*)(p + 16);
    v16h out;
#pragma unroll
    for (int i = 0; i < 8; ++i) { out[i] = lo[i]; out[i + 8] = hi[i]; }
    return out;
}

// B-fragment (32x16 f16, K x N): lane n=lane&15 holds column N=n; lanes 0-15 K={0..7, 16..23},
// lanes 16-31 K={8..15, 24..31} (same K order as the A-fragment).
__device__ __forceinline__ v16h load_b_frag(const _Float16* colbase, int ldh, int kb, int lane) {
    int n = lane & 15;
    const _Float16* p = colbase + (size_t)n * ldh + kb + ((lane & 16) ? 8 : 0);
    v8h lo = *(const v8h*)(p);
    v8h hi = *(const v8h*)(p + 16);
    v16h out;
#pragma unroll
    for (int i = 0; i < 8; ++i) { out[i] = lo[i]; out[i + 8] = hi[i]; }
    return out;
}

__device__ __forceinline__ v8f wmma_f16(v16h a, v16h b, v8f c) {
    return __builtin_amdgcn_wmma_f32_16x16x32_f16(false, a, false, b, (short)0, c, false, false);
}

// ---- gfx1250 async global->LDS staging (ASYNCcnt path, inline asm) ----
// LDS byte address == low 32 bits of the generic pointer (flat LDS aperture keeps
// the offset in addr[31:0]).  Each lane copies 16B chunks; drained with s_wait_asynccnt.
template<int CHUNKS_PER_THREAD>
__device__ __forceinline__ void stage_async_b128(const _Float16* __restrict__ g,
                                                 _Float16* __restrict__ s, int tid) {
#pragma unroll
    for (int i = 0; i < CHUNKS_PER_THREAD; ++i) {
        int c = (i * 128 + tid) * 8;                       // halves
        unsigned lds_addr = (unsigned)(size_t)(s + c);
        unsigned long long gaddr = (unsigned long long)(size_t)(g + c);
        asm volatile("global_load_async_to_lds_b128 %0, %1, off"
                     :: "v"(lds_addr), "v"(gaddr) : "memory");
    }
    asm volatile("s_wait_asynccnt 0" ::: "memory");
}

// ---------------- generic small kernels ----------------
__global__ void zero_f(float* p, int n) {
    int i = blockIdx.x * blockDim.x + threadIdx.x;
    if (i < n) p[i] = 0.0f;
}

// unfold with 3x3 / pad 1: src (512, C, H, H) -> patch (512, C*9, H*H)
template<int C, int H>
__global__ void unfold3(const float* __restrict__ src, float* __restrict__ patch) {
    int i = blockIdx.x * blockDim.x + threadIdx.x;
    const int HW = H * H;
    int total = 512 * C * 9 * HW;
    if (i >= total) return;
    int hw = i % HW;
    int t  = i / HW;
    int k9 = t % 9;
    int c  = (t / 9) % C;
    int n  = t / (9 * C);
    int ki = k9 / 3, kj = k9 % 3;
    int ii = hw / H + ki - 1, jj = hw % H + kj - 1;
    float v = 0.0f;
    if (ii >= 0 && ii < H && jj >= 0 && jj < H)
        v = src[((size_t)n * C + c) * HW + ii * H + jj];
    patch[i] = v;
}

// pack flat rows (E,D) f32 -> (E,DP) f16 (zero padded) + row squared norms
template<int D, int DP>
__global__ void pack_rows(const float* __restrict__ src, _Float16* __restrict__ dst,
                          float* __restrict__ anorm, int E) {
    int e = blockIdx.x * blockDim.x + threadIdx.x;
    if (e >= E) return;
    float s = 0.0f;
    const float* r = src + (size_t)e * D;
    _Float16* w = dst + (size_t)e * DP;
#pragma unroll
    for (int d = 0; d < D; ++d) { float v = r[d]; s += v * v; w[d] = (_Float16)v; }
#pragma unroll
    for (int d = D; d < DP; ++d) w[d] = (_Float16)0.0f;
    anorm[e] = s;
}

template<int D>
__global__ void gather_centers(const float* __restrict__ flat, const int* __restrict__ idx,
                               float* __restrict__ centers) {
    int i = blockIdx.x * blockDim.x + threadIdx.x;
    if (i >= 128 * D) return;
    int k = i / D, d = i % D;
    centers[i] = flat[(size_t)idx[k] * D + d];
}

// f32 centers -> f16 (128,DP) row-major, f16 transposed (TROWS,128), squared norms
template<int D, int DP, int TROWS>
__global__ void pack_centers(const float* __restrict__ centers, _Float16* __restrict__ ch,
                             _Float16* __restrict__ cth, float* __restrict__ cnorm) {
    int k = threadIdx.x;
    if (k >= 128) return;
    float s = 0.0f;
#pragma unroll 4
    for (int d = 0; d < D; ++d) {
        float v = centers[(size_t)k * D + d];
        s += v * v;
        ch[(size_t)k * DP + d] = (_Float16)v;
    }
    for (int d = D; d < DP; ++d) ch[(size_t)k * DP + d] = (_Float16)0.0f;
    for (int d = 0; d < TROWS; ++d)
        cth[(size_t)d * 128 + k] = (d < D) ? (_Float16)centers[(size_t)k * D + d] : (_Float16)0.0f;
    cnorm[k] = s;
}

template<int D>
__global__ void centers_update(float* __restrict__ centers, const float* __restrict__ sums,
                               const float* __restrict__ cnt) {
    int i = blockIdx.x * blockDim.x + threadIdx.x;
    if (i >= 128 * D) return;
    int k = i / D;
    float c = cnt[k];
    if (c > 0.0f) centers[i] = sums[i] / fmaxf(c, 1.0f);
}

// ---------------- k-means assignment + segment accumulate (WMMA) ----------------
// Block stages the full 128xDP f16 center matrix into LDS via async b128 copies,
// then each wave computes a 16-row x 128-center tile with WMMA.
template<int D, int DP>
__global__ void kmeans_assign(const _Float16* __restrict__ fH, const float* __restrict__ fF,
                              const _Float16* __restrict__ cH, const float* __restrict__ cnorm,
                              float* __restrict__ sums, float* __restrict__ cnt, int E) {
    __shared__ __align__(16) _Float16 scen[128 * DP];
    int tid   = threadIdx.x;
    int gwave = (blockIdx.x * blockDim.x + tid) >> 5;
    int lane  = tid & 31;
    int row0  = gwave << 4;

    stage_async_b128<DP / 8>(cH, scen, tid);   // 128*DP halves, 16B chunks, 128 threads
    __syncthreads();

    v8f acc[8];
#pragma unroll
    for (int t = 0; t < 8; ++t)
#pragma unroll
        for (int i = 0; i < 8; ++i) acc[t][i] = 0.0f;

    const _Float16* aBase = fH + (size_t)row0 * DP;
#pragma unroll
    for (int kb = 0; kb < DP; kb += 32) {
        v16h a = load_a_frag(aBase, DP, kb, lane);
#pragma unroll
        for (int t = 0; t < 8; ++t) {
            v16h b = load_b_frag(scen + (size_t)(t * 16) * DP, DP, kb, lane);
            acc[t] = wmma_f16(a, b, acc[t]);
        }
    }

    int col  = lane & 15;
    int rsel = (lane & 16) ? 8 : 0;
#pragma unroll
    for (int r = 0; r < 8; ++r) {
        float best = 3.4e38f;
        int   bidx = 0;
#pragma unroll
        for (int t = 0; t < 8; ++t) {
            float v = cnorm[t * 16 + col] - 2.0f * acc[t][r];
            if (v < best) { best = v; bidx = t * 16 + col; }
        }
        // min-reduce inside each 16-lane half (rows r and r+8); first-index tie break
#pragma unroll
        for (int m = 1; m < 16; m <<= 1) {
            float ov = __shfl_xor(best, m, 32);
            int   oi = __shfl_xor(bidx, m, 32);
            if (ov < best || (ov == best && oi < bidx)) { best = ov; bidx = oi; }
        }
        int row = row0 + r + rsel;
        for (int d = col; d < D; d += 16)
            atomicAdd(&sums[(size_t)bidx * D + d], fF[(size_t)row * D + d]);
        if (col == 0) atomicAdd(&cnt[bidx], 1.0f);
    }
}

// ---------------- soft quantization (WMMA x2) ----------------
// dist = sqrt(max(|a|^2-2ab+|b|^2,0)); soft = softmax(-dist*1e-5); trans = soft@centers;
// final = T/(T+1)*trans + 1/(T+1)*flat   (in place on fF).
template<int D, int DP, int NT>
__global__ void soft_transform(const _Float16* __restrict__ fH, float* __restrict__ fF,
                               const _Float16* __restrict__ cH, const _Float16* __restrict__ cTh,
                               const float* __restrict__ cnorm, const float* __restrict__ anorm,
                               int E) {
    __shared__ __align__(16) _Float16 scen[128 * DP];
    __shared__ __align__(16) _Float16 lsoft[4 * 16 * 128];   // 4 waves/block
    int tid   = threadIdx.x;
    int lwave = tid >> 5;
    int gwave = (blockIdx.x * blockDim.x + tid) >> 5;
    int lane  = tid & 31;
    int row0  = gwave << 4;

    stage_async_b128<DP / 8>(cH, scen, tid);
    __syncthreads();

    v8f acc[8];
#pragma unroll
    for (int t = 0; t < 8; ++t)
#pragma unroll
        for (int i = 0; i < 8; ++i) acc[t][i] = 0.0f;

    const _Float16* aBase = fH + (size_t)row0 * DP;
#pragma unroll
    for (int kb = 0; kb < DP; kb += 32) {
        v16h a = load_a_frag(aBase, DP, kb, lane);
#pragma unroll
        for (int t = 0; t < 8; ++t) {
            v16h b = load_b_frag(scen + (size_t)(t * 16) * DP, DP, kb, lane);
            acc[t] = wmma_f16(a, b, acc[t]);
        }
    }

    int col  = lane & 15;
    int rsel = (lane & 16) ? 8 : 0;
    _Float16* ls = lsoft + lwave * 16 * 128;
#pragma unroll
    for (int r = 0; r < 8; ++r) {
        int   row = row0 + r + rsel;
        float an  = anorm[row];
        float lg[8];
        float mx = -3.4e38f;
#pragma unroll
        for (int t = 0; t < 8; ++t) {
            float d2   = an - 2.0f * acc[t][r] + cnorm[t * 16 + col];
            float dist = sqrtf(fmaxf(d2, 0.0f));
            lg[t] = -dist * 1e-5f;
            mx = fmaxf(mx, lg[t]);
        }
#pragma unroll
        for (int m = 1; m < 16; m <<= 1) mx = fmaxf(mx, __shfl_xor(mx, m, 32));
        float se = 0.0f;
#pragma unroll
        for (int t = 0; t < 8; ++t) { lg[t] = __expf(lg[t] - mx); se += lg[t]; }
#pragma unroll
        for (int m = 1; m < 16; m <<= 1) se += __shfl_xor(se, m, 32);
        float inv = 1.0f / se;
#pragma unroll
        for (int t = 0; t < 8; ++t)
            ls[(r + rsel) * 128 + t * 16 + col] = (_Float16)(lg[t] * inv);
    }
    __syncthreads();

    // stage 2: trans = soft (16x128) @ centers (128xD) — A from LDS, B = transposed centers
    v8f tacc[NT];
#pragma unroll
    for (int t = 0; t < NT; ++t)
#pragma unroll
        for (int i = 0; i < 8; ++i) tacc[t][i] = 0.0f;
#pragma unroll
    for (int kb = 0; kb < 128; kb += 32) {
        v16h a = load_a_frag(ls, 128, kb, lane);
#pragma unroll
        for (int t = 0; t < NT; ++t) {
            v16h b = load_b_frag(cTh + (size_t)(t * 16) * 128, 128, kb, lane);
            tacc[t] = wmma_f16(a, b, tacc[t]);
        }
    }
    const float A_ = 1e-5f / (1.0f + 1e-5f);
    const float B_ = 1.0f / (1.0f + 1e-5f);
#pragma unroll
    for (int r = 0; r < 8; ++r) {
        int row = row0 + r + rsel;
#pragma unroll
        for (int t = 0; t < NT; ++t) {
            int d = t * 16 + col;
            if (d < D) {
                size_t ix = (size_t)row * D + d;
                fF[ix] = A_ * tacc[t][r] + B_ * fF[ix];
            }
        }
    }
}

// ---------------- fused fold + stride-3 conv + bias + relu + 2x2 pool ----------------
// fold/stride-3-conv is a bijection patch<->pixel with edge cut at (i==0&&ki==0)/(j==0&&kj==0).
template<int CIN, int HH, int COUT>
__global__ void conv_relu_pool(const float* __restrict__ fin, const float* __restrict__ w,
                               const float* __restrict__ bias, float* __restrict__ out) {
    const int P  = HH / 2;
    const int HW = HH * HH;
    int i = blockIdx.x * blockDim.x + threadIdx.x;
    int total = 512 * COUT * P * P;
    if (i >= total) return;
    int pj = i % P, pi = (i / P) % P, o = (i / (P * P)) % COUT, n = i / (P * P * COUT);
    const float* wB = w + (size_t)o * CIN * 9;
    float m = -3.4e38f;
#pragma unroll
    for (int di = 0; di < 2; ++di)
#pragma unroll
        for (int dj = 0; dj < 2; ++dj) {
            int ii = 2 * pi + di, jj = 2 * pj + dj;
            float s = bias[o];
            for (int c = 0; c < CIN; ++c)
#pragma unroll
                for (int ki = 0; ki < 3; ++ki)
#pragma unroll
                    for (int kj = 0; kj < 3; ++kj) {
                        if ((ii == 0 && ki == 0) || (jj == 0 && kj == 0)) continue;
                        s += wB[c * 9 + ki * 3 + kj] *
                             fin[((size_t)n * CIN * 9 + c * 9 + ki * 3 + kj) * HW + ii * HH + jj];
                    }
            m = fmaxf(m, s);
        }
    out[i] = fmaxf(m, 0.0f);
}

__global__ void fc_kernel(const float* __restrict__ a, const float* __restrict__ w,
                          const float* __restrict__ b, float* __restrict__ out) {
    int i = blockIdx.x * blockDim.x + threadIdx.x;
    if (i >= 512 * 10) return;
    int o = i % 10, n = i / 10;
    float s = b[o];
    const float* ar = a + (size_t)n * 1568;
    const float* wr = w + (size_t)o * 1568;
#pragma unroll 8
    for (int k = 0; k < 1568; ++k) s += ar[k] * wr[k];
    out[i] = s;
}

// ---------------- host orchestration ----------------
extern "C" void kernel_launch(void* const* d_in, const int* in_sizes, int n_in,
                              void* d_out, int out_size, void* d_ws, size_t ws_size,
                              hipStream_t stream) {
    const float* x   = (const float*)d_in[0];
    const float* w1  = (const float*)d_in[1];
    const float* b1  = (const float*)d_in[2];
    const float* w2  = (const float*)d_in[3];
    const float* b2  = (const float*)d_in[4];
    const float* fw  = (const float*)d_in[5];
    const float* fb  = (const float*)d_in[6];
    const int*   id1 = (const int*)d_in[7];
    const int*   id2 = (const int*)d_in[8];

    const int E1 = 512 * 784;   // 401408, D=9,  DP=32
    const int E2 = 512 * 196;   // 100352, D=144, DP=160

    char* ws = (char*)d_ws;
    size_t off = 0;
    auto alloc = [&](size_t bytes) -> void* {
        void* p = ws + off;
        off += (bytes + 255) & ~(size_t)255;
        return p;
    };

    float* act1 = (float*)alloc((size_t)512 * 16 * 14 * 14 * 4);
    float* act2 = (float*)alloc((size_t)512 * 32 * 7 * 7 * 4);
    size_t arena = off;

    const int B = 256;
    auto g = [](int n, int b) { return (n + b - 1) / b; };

    // =============== layer 1 (C=1, H=28, D=9, DP=32) ===============
    {
        off = arena;
        float*    patch = (float*)   alloc((size_t)E1 * 9 * 4);
        _Float16* ph    = (_Float16*)alloc((size_t)E1 * 32 * 2);
        float*    an    = (float*)   alloc((size_t)E1 * 4);
        float*    cen   = (float*)   alloc(128 * 9 * 4);
        _Float16* chh   = (_Float16*)alloc(128 * 32 * 2);
        _Float16* cth   = (_Float16*)alloc(16 * 128 * 2);
        float*    cn    = (float*)   alloc(128 * 4);
        float*    sums  = (float*)   alloc((128 * 9 + 128) * 4);
        float*    cnt   = sums + 128 * 9;

        unfold3<1, 28><<<g(512 * 9 * 784, B), B, 0, stream>>>(x, patch);
        pack_rows<9, 32><<<g(E1, B), B, 0, stream>>>(patch, ph, an, E1);
        gather_centers<9><<<g(128 * 9, B), B, 0, stream>>>(patch, id1, cen);
        for (int it = 0; it < 10; ++it) {
            pack_centers<9, 32, 16><<<1, 128, 0, stream>>>(cen, chh, cth, cn);
            zero_f<<<g(128 * 9 + 128, B), B, 0, stream>>>(sums, 128 * 9 + 128);
            kmeans_assign<9, 32><<<E1 / 64, 128, 0, stream>>>(ph, patch, chh, cn, sums, cnt, E1);
            centers_update<9><<<g(128 * 9, B), B, 0, stream>>>(cen, sums, cnt);
        }
        pack_centers<9, 32, 16><<<1, 128, 0, stream>>>(cen, chh, cth, cn);
        soft_transform<9, 32, 1><<<E1 / 64, 128, 0, stream>>>(ph, patch, chh, cth, cn, an, E1);
        conv_relu_pool<1, 28, 16><<<g(512 * 16 * 14 * 14, B), B, 0, stream>>>(patch, w1, b1, act1);
    }

    // =============== layer 2 (C=16, H=14, D=144, DP=160) ===============
    {
        off = arena;
        float*    patch = (float*)   alloc((size_t)E2 * 144 * 4);
        _Float16* ph    = (_Float16*)alloc((size_t)E2 * 160 * 2);
        float*    an    = (float*)   alloc((size_t)E2 * 4);
        float*    cen   = (float*)   alloc(128 * 144 * 4);
        _Float16* chh   = (_Float16*)alloc(128 * 160 * 2);
        _Float16* cth   = (_Float16*)alloc(144 * 128 * 2);
        float*    cn    = (float*)   alloc(128 * 4);
        float*    sums  = (float*)   alloc((128 * 144 + 128) * 4);
        float*    cnt   = sums + 128 * 144;

        unfold3<16, 14><<<g(512 * 144 * 196, B), B, 0, stream>>>(act1, patch);
        pack_rows<144, 160><<<g(E2, B), B, 0, stream>>>(patch, ph, an, E2);
        gather_centers<144><<<g(128 * 144, B), B, 0, stream>>>(patch, id2, cen);
        for (int it = 0; it < 10; ++it) {
            pack_centers<144, 160, 144><<<1, 128, 0, stream>>>(cen, chh, cth, cn);
            zero_f<<<g(128 * 144 + 128, B), B, 0, stream>>>(sums, 128 * 144 + 128);
            kmeans_assign<144, 160><<<E2 / 64, 128, 0, stream>>>(ph, patch, chh, cn, sums, cnt, E2);
            centers_update<144><<<g(128 * 144, B), B, 0, stream>>>(cen, sums, cnt);
        }
        pack_centers<144, 160, 144><<<1, 128, 0, stream>>>(cen, chh, cth, cn);
        soft_transform<144, 160, 9><<<E2 / 64, 128, 0, stream>>>(ph, patch, chh, cth, cn, an, E2);
        conv_relu_pool<16, 14, 32><<<g(512 * 32 * 7 * 7, B), B, 0, stream>>>(patch, w2, b2, act2);
    }

    fc_kernel<<<g(512 * 10, B), B, 0, stream>>>(act2, fw, fb, (float*)d_out);
}
